// NodeModel_39444979647200
// MI455X (gfx1250) — matrix operand order repaired and checked by reading, not compiled, not run on hardware
//
#include <hip/hip_runtime.h>
#include <hip/hip_bf16.h>
#include <stdint.h>

#define N_NODES 50000
#define N_EDGES 800000
#define N_GLOB  256
#define EDGE_TILES 4   // 128-edge tiles per block (amortizes LDS weight staging)

typedef __attribute__((ext_vector_type(16))) _Float16 v16h;
typedef __attribute__((ext_vector_type(8)))  _Float16 v8h;
typedef __attribute__((ext_vector_type(8)))  float    v8f;

union V16U { v16h v; v8h h[2]; };

// ---------- WMMA fragment loaders (CDNA5 wave32 layouts, CDNA5 ISA) ----------

// A: 16x32 f16, row-major source tile in LDS (ld halves per row).
// lanes 0-15: row M=lane, K {k..k+7, k+16..k+23}; lanes 16-31: K {k+8..k+15, k+24..k+31}
__device__ __forceinline__ v16h load_a_frag(const _Float16* base, int lda, int k, int lane) {
  const _Float16* p = base + (lane & 15) * lda + k + ((lane >> 4) << 3);
  V16U u;
  u.h[0] = *(const v8h*)p;
  u.h[1] = *(const v8h*)(p + 16);
  return u.v;
}

// B: 32x16 f16. wT is [16 out-cols][kd in] row-major (pre-transposed weights).
// lane n<16: column n, K {k..k+7, k+16..k+23}; lane n+16: column n, K {k+8..k+15, k+24..k+31}.
__device__ __forceinline__ v16h load_b_frag(const _Float16* wT, int kd, int k, int lane) {
  const _Float16* p = wT + (lane & 15) * kd + k + ((lane >> 4) << 3);
  V16U u;
  u.h[0] = *(const v8h*)p;
  u.h[1] = *(const v8h*)(p + 16);
  return u.v;
}

// 16xKD @ KDx64 -> 16x64 in four 16x16 f32 accumulators.
template <int KD>
__device__ __forceinline__ void gemm_tile(const _Float16* aBase, int lda,
                                          const _Float16* wT, v8f* acc, int lane) {
  v8f z = {0.f, 0.f, 0.f, 0.f, 0.f, 0.f, 0.f, 0.f};
  #pragma unroll
  for (int nt = 0; nt < 4; ++nt) acc[nt] = z;
  #pragma unroll
  for (int k = 0; k < KD; k += 32) {
    v16h a = load_a_frag(aBase, lda, k, lane);
    #pragma unroll
    for (int nt = 0; nt < 4; ++nt) {
      v16h b = load_b_frag(wT + nt * 16 * KD, KD, k, lane);
      acc[nt] = __builtin_amdgcn_wmma_f32_16x16x32_f16(
          false, a, false, b, (short)0, acc[nt], false, false);
    }
  }
}

// bias + ReLU + store f16 to [16][64] LDS tile (accumulator layout -> row-major)
__device__ __forceinline__ void store_relu_h(const v8f* acc, const float* __restrict__ bias,
                                             _Float16* outp, int lane) {
  const int gq = lane >> 4, nl = lane & 15;
  #pragma unroll
  for (int nt = 0; nt < 4; ++nt) {
    const int Nn = nl + nt * 16;
    const float bv = bias[Nn];
    #pragma unroll
    for (int r = 0; r < 8; ++r) {
      const int M = r + 8 * gq;
      outp[M * 64 + Nn] = (_Float16)fmaxf(acc[nt][r] + bv, 0.f);
    }
  }
}

// bias + ReLU + store f32 to [16][64] scratch (for LayerNorm)
__device__ __forceinline__ void store_relu_f(const v8f* acc, const float* __restrict__ bias,
                                             float* outp, int lane) {
  const int gq = lane >> 4, nl = lane & 15;
  #pragma unroll
  for (int nt = 0; nt < 4; ++nt) {
    const int Nn = nl + nt * 16;
    const float bv = bias[Nn];
    #pragma unroll
    for (int r = 0; r < 8; ++r) {
      const int M = r + 8 * gq;
      outp[M * 64 + Nn] = fmaxf(acc[nt][r] + bv, 0.f);
    }
  }
}

// LayerNorm over 64 feats: f32 scratch [16][64] -> f16 tile [16][64]
__device__ __forceinline__ void layernorm_tile(const float* lnbuf, _Float16* outp,
                                               const float* __restrict__ gam,
                                               const float* __restrict__ bet, int lane) {
  const int row = lane & 15, half = lane >> 4;
  const float* rp = lnbuf + row * 64 + half * 32;
  float s = 0.f, s2 = 0.f;
  #pragma unroll
  for (int c = 0; c < 32; ++c) { float v = rp[c]; s += v; s2 += v * v; }
  s  += __shfl_xor(s, 16, 32);
  s2 += __shfl_xor(s2, 16, 32);
  const float mu   = s * (1.f / 64.f);
  const float var  = s2 * (1.f / 64.f) - mu * mu;
  const float rstd = rsqrtf(var + 1e-5f);
  _Float16* hp = outp + row * 64 + half * 32;
  #pragma unroll
  for (int c = 0; c < 32; ++c) {
    const int f = half * 32 + c;
    hp[c] = (_Float16)((rp[c] - mu) * rstd * gam[f] + bet[f]);
  }
}

__device__ __forceinline__ void cvt_store4(_Float16* dst, float4 v) {
  union { _Float16 h[4]; uint2 u; } p;
  p.h[0] = (_Float16)v.x; p.h[1] = (_Float16)v.y;
  p.h[2] = (_Float16)v.z; p.h[3] = (_Float16)v.w;
  *(uint2*)dst = p.u;
}

// Async global->LDS stage of packed f16 weights (pure byte copy -> ASYNCcnt path,
// no VGPR round-trip). halves must be a multiple of 8*256.
__device__ __forceinline__ void stage_weights_async(_Float16* dstLds,
                                                    const _Float16* __restrict__ src,
                                                    int halves, int tid) {
  const int n128 = halves >> 3;               // 16-byte chunks
  for (int i = tid; i < n128; i += 256) {
    const unsigned lds = (unsigned)(uintptr_t)(dstLds + i * 8);  // LDS byte offset
    const unsigned off = (unsigned)(i * 16);                      // global byte offset
    asm volatile("global_load_async_to_lds_b128 %0, %1, %2"
                 :: "v"(lds), "v"(off), "s"(src)
                 : "memory");
  }
  asm volatile("s_wait_asynccnt 0x0" ::: "memory");
}

// ---------------------------- helper kernels ----------------------------

__global__ void zero_kernel(float* p, int n) {
  int i = blockIdx.x * blockDim.x + threadIdx.x;
  if (i < n) p[i] = 0.f;
}

// Build transposed f16 weights: wt1 = [wT0 64x128 | wT1 64x64 | wT2 64x64 | wTo 64x64]
//                               wt2 = [wT0 64x192 | wT1 64x64 | wT2 64x64 | wTo 64x64]
__global__ void pack_weights(const float* __restrict__ w10, const float* __restrict__ w11,
                             const float* __restrict__ w12, const float* __restrict__ w1o,
                             const float* __restrict__ w20, const float* __restrict__ w21,
                             const float* __restrict__ w22, const float* __restrict__ w2o,
                             _Float16* wt1, _Float16* wt2) {
  int i = blockIdx.x * blockDim.x + threadIdx.x;
  if (i < 20480) {
    _Float16 v;
    if (i < 8192)        { int o = i >> 7,  k = i & 127;          v = (_Float16)w10[k * 64 + o]; }
    else if (i < 12288)  { int j = i - 8192;  int o = j >> 6, k = j & 63; v = (_Float16)w11[k * 64 + o]; }
    else if (i < 16384)  { int j = i - 12288; int o = j >> 6, k = j & 63; v = (_Float16)w12[k * 64 + o]; }
    else                 { int j = i - 16384; int o = j >> 6, k = j & 63; v = (_Float16)w1o[k * 64 + o]; }
    wt1[i] = v;
  }
  if (i < 24576) {
    _Float16 v;
    if (i < 12288)       { int o = i / 192, k = i % 192;          v = (_Float16)w20[k * 64 + o]; }
    else if (i < 16384)  { int j = i - 12288; int o = j >> 6, k = j & 63; v = (_Float16)w21[k * 64 + o]; }
    else if (i < 20480)  { int j = i - 16384; int o = j >> 6, k = j & 63; v = (_Float16)w22[k * 64 + o]; }
    else                 { int j = i - 20480; int o = j >> 6, k = j & 63; v = (_Float16)w2o[k * 64 + o]; }
    wt2[i] = v;
  }
}

// --------------------------- edge MLP + scatter ---------------------------
// block = 256 threads (8 waves); EDGE_TILES tiles of 128 edges; 16 edges per wave.
__global__ __launch_bounds__(256) void edge_kernel(
    const float* __restrict__ x, const int* __restrict__ ei,
    const float* __restrict__ ea, const _Float16* __restrict__ wt1,
    const float* __restrict__ b0, const float* __restrict__ b1,
    const float* __restrict__ b2, const float* __restrict__ gam,
    const float* __restrict__ bet, const float* __restrict__ bo,
    float* __restrict__ agg) {
  __shared__ __align__(16) _Float16 sW[20480];         // 40 KB packed weights
  __shared__ __align__(16) _Float16 sA[128 * 128];     // 32 KB msg tile (f16) / LN scratch (f32)
  __shared__ __align__(16) _Float16 sH[128 * 64];      // 16 KB hidden tile
  const int tid = threadIdx.x;
  const int lane = tid & 31;
  const int wave = tid >> 5;

  stage_weights_async(sW, wt1, 20480, tid);            // ASYNCcnt path

  const _Float16* W0 = sW;           // 64x128
  const _Float16* W1 = sW + 8192;    // 64x64
  const _Float16* W2 = sW + 12288;   // 64x64
  const _Float16* Wo = sW + 16384;   // 64x64
  const int rb = wave * 16;
  v8f acc[4];

  const int ebase = blockIdx.x * (128 * EDGE_TILES);
  for (int t = 0; t < EDGE_TILES; ++t) {
    const int e0 = ebase + t * 128;
    // stage msg = [x[row] | edge_attr] as f16 tile [128][128]
    {
      const int el = tid & 127;
      const int part = tid >> 7;                 // 0: x[row[e]], 1: edge_attr[e]
      const int e = e0 + el;
      _Float16* dst = sA + el * 128 + part * 64;
      if (e < N_EDGES) {
        const float* src = part ? (ea + (size_t)e * 64)
                                : (x + (size_t)ei[e] * 64);
        if (part && t + 1 < EDGE_TILES && e + 128 < N_EDGES)
          __builtin_prefetch(ea + (size_t)(e + 128) * 64, 0, 0);  // global_prefetch_b8
        #pragma unroll
        for (int i = 0; i < 16; ++i) cvt_store4(dst + i * 4, ((const float4*)src)[i]);
      } else {
        uint2 zz = {0u, 0u};
        #pragma unroll
        for (int i = 0; i < 16; ++i) *(uint2*)(dst + i * 4) = zz;
      }
    }
    __syncthreads();

    gemm_tile<128>(sA + rb * 128, 128, W0, acc, lane);
    store_relu_h(acc, b0, sH + rb * 64, lane);
    __syncthreads();

    gemm_tile<64>(sH + rb * 64, 64, W1, acc, lane);
    store_relu_h(acc, b1, sH + rb * 64, lane);
    __syncthreads();

    gemm_tile<64>(sH + rb * 64, 64, W2, acc, lane);
    float* lnbuf = (float*)(sA + rb * 128);      // reuse this wave's msg rows as f32 scratch
    store_relu_f(acc, b2, lnbuf, lane);
    __syncthreads();

    layernorm_tile(lnbuf, sH + rb * 64, gam, bet, lane);
    __syncthreads();                             // after this, sA is free for next tile

    gemm_tile<64>(sH + rb * 64, 64, Wo, acc, lane);

    // scatter-sum into agg[col[e]] (L2-resident, no-return f32 atomics)
    {
      const int gq = lane >> 4, nl = lane & 15;
      #pragma unroll
      for (int nt = 0; nt < 4; ++nt) {
        const int Nn = nl + nt * 16;
        const float bv = bo[Nn];
        #pragma unroll
        for (int r = 0; r < 8; ++r) {
          const int M = r + 8 * gq;
          const int e = e0 + rb + M;
          if (e < N_EDGES) {
            const int c = ei[N_EDGES + e];
            atomicAdd(&agg[(size_t)c * 64 + Nn], acc[nt][r] + bv);
          }
        }
      }
    }
  }
}

// --------------------------- node MLP + residual ---------------------------
__global__ __launch_bounds__(256) void node_kernel(
    const float* __restrict__ x, const float* __restrict__ agg,
    const float* __restrict__ u, const int* __restrict__ batch,
    const _Float16* __restrict__ wt2,
    const float* __restrict__ b0, const float* __restrict__ b1,
    const float* __restrict__ b2, const float* __restrict__ gam,
    const float* __restrict__ bet, const float* __restrict__ bo,
    float* __restrict__ out) {
  __shared__ __align__(16) _Float16 sW[24576];         // 48 KB packed weights
  __shared__ __align__(16) _Float16 sA[128 * 192];     // 48 KB input tile / LN scratch
  __shared__ __align__(16) _Float16 sH[128 * 64];      // 16 KB hidden tile
  const int tid = threadIdx.x;
  const int lane = tid & 31;
  const int wave = tid >> 5;

  stage_weights_async(sW, wt2, 24576, tid);            // ASYNCcnt path

  const int n0 = blockIdx.x * 128;
  {
    const int nl = tid & 127;
    const int seg = tid >> 7;                  // 0: feats 0-31, 1: feats 32-63
    const int n = n0 + nl;
    #pragma unroll
    for (int srci = 0; srci < 3; ++srci) {     // [x | agg | u[batch]]
      _Float16* dst = sA + nl * 192 + srci * 64 + seg * 32;
      if (n < N_NODES) {
        const float* src;
        if (srci == 0)      src = x   + (size_t)n * 64;
        else if (srci == 1) src = agg + (size_t)n * 64;
        else                src = u   + (size_t)batch[n] * 64;
        src += seg * 32;
        #pragma unroll
        for (int i = 0; i < 8; ++i) cvt_store4(dst + i * 4, ((const float4*)src)[i]);
      } else {
        uint2 zz = {0u, 0u};
        #pragma unroll
        for (int i = 0; i < 8; ++i) *(uint2*)(dst + i * 4) = zz;
      }
    }
  }
  __syncthreads();

  const _Float16* W0 = sW;           // 64x192
  const _Float16* W1 = sW + 12288;   // 64x64
  const _Float16* W2 = sW + 16384;   // 64x64
  const _Float16* Wo = sW + 20480;   // 64x64
  const int rb = wave * 16;
  v8f acc[4];

  gemm_tile<192>(sA + rb * 192, 192, W0, acc, lane);
  store_relu_h(acc, b0, sH + rb * 64, lane);
  __syncthreads();

  gemm_tile<64>(sH + rb * 64, 64, W1, acc, lane);
  store_relu_h(acc, b1, sH + rb * 64, lane);
  __syncthreads();

  gemm_tile<64>(sH + rb * 64, 64, W2, acc, lane);
  float* lnbuf = (float*)(sA + rb * 192);
  store_relu_f(acc, b2, lnbuf, lane);
  __syncthreads();

  layernorm_tile(lnbuf, sH + rb * 64, gam, bet, lane);
  __syncthreads();

  gemm_tile<64>(sH + rb * 64, 64, Wo, acc, lane);

  // out = x + mlp2(...)
  {
    const int gq = lane >> 4, nl = lane & 15;
    #pragma unroll
    for (int nt = 0; nt < 4; ++nt) {
      const int Nn = nl + nt * 16;
      const float bv = bo[Nn];
      #pragma unroll
      for (int r = 0; r < 8; ++r) {
        const int M = r + 8 * gq;
        const int n = n0 + rb + M;
        if (n < N_NODES)
          out[(size_t)n * 64 + Nn] = x[(size_t)n * 64 + Nn] + acc[nt][r] + bv;
      }
    }
  }
}

// ------------------------------- launcher -------------------------------

extern "C" void kernel_launch(void* const* d_in, const int* in_sizes, int n_in,
                              void* d_out, int out_size, void* d_ws, size_t ws_size,
                              hipStream_t stream) {
  const float* x     = (const float*)d_in[0];
  const int*   ei    = (const int*)d_in[1];
  const float* ea    = (const float*)d_in[2];
  const float* u     = (const float*)d_in[3];
  const int*   batch = (const int*)d_in[4];
  const float* w1_0 = (const float*)d_in[5];
  const float* b1_0 = (const float*)d_in[6];
  const float* w1_1 = (const float*)d_in[7];
  const float* b1_1 = (const float*)d_in[8];
  const float* w1_2 = (const float*)d_in[9];
  const float* b1_2 = (const float*)d_in[10];
  const float* g1   = (const float*)d_in[11];
  const float* be1  = (const float*)d_in[12];
  const float* w1_o = (const float*)d_in[13];
  const float* b1_o = (const float*)d_in[14];
  const float* w2_0 = (const float*)d_in[15];
  const float* b2_0 = (const float*)d_in[16];
  const float* w2_1 = (const float*)d_in[17];
  const float* b2_1 = (const float*)d_in[18];
  const float* w2_2 = (const float*)d_in[19];
  const float* b2_2 = (const float*)d_in[20];
  const float* g2   = (const float*)d_in[21];
  const float* be2  = (const float*)d_in[22];
  const float* w2_o = (const float*)d_in[23];
  const float* b2_o = (const float*)d_in[24];

  float*    agg = (float*)d_ws;                                   // N*64 f32 = 12.8 MB
  _Float16* wt1 = (_Float16*)((char*)d_ws + (size_t)N_NODES * 64 * sizeof(float));
  _Float16* wt2 = wt1 + 20480;

  zero_kernel<<<(N_NODES * 64 + 255) / 256, 256, 0, stream>>>(agg, N_NODES * 64);
  pack_weights<<<96, 256, 0, stream>>>(w1_0, w1_1, w1_2, w1_o,
                                       w2_0, w2_1, w2_2, w2_o, wt1, wt2);
  const int edges_per_block = 128 * EDGE_TILES;
  edge_kernel<<<(N_EDGES + edges_per_block - 1) / edges_per_block, 256, 0, stream>>>(
      x, ei, ea, wt1, b1_0, b1_1, b1_2, g1, be1, b1_o, agg);
  node_kernel<<<(N_NODES + 127) / 128, 256, 0, stream>>>(
      x, agg, u, batch, wt2, b2_0, b2_1, b2_2, g2, be2, b2_o, (float*)d_out);
}
